// RNN_72456098283796
// MI455X (gfx1250) — matrix operand order repaired and checked by reading, not compiled, not run on hardware
//
#include <hip/hip_runtime.h>
#include <hip/hip_bf16.h>

#define VOCAB  1024
#define HIDDEN 1024
#define BATCH  64
#define SEQ    512

typedef __attribute__((ext_vector_type(16))) __bf16 v16bf;
typedef __attribute__((ext_vector_type(8)))  __bf16 v8bf;
typedef __attribute__((ext_vector_type(8)))  float  v8f;
typedef __attribute__((ext_vector_type(4)))  unsigned int u32x4;
typedef __attribute__((ext_vector_type(8)))  int  i32x8;
typedef __attribute__((ext_vector_type(4)))  int  i32x4;

union ABu  { v16bf v; v8bf h[2]; };
union Accu { v8f v; float f[8]; };

__device__ __forceinline__ v8f wmma_bf16(v16bf a, v16bf b, v8f c) {
    // v_wmma_f32_16x16x32_bf16 (probe-confirmed 8-arg signature)
    return __builtin_amdgcn_wmma_f32_16x16x32_bf16(
        false, a, false, b, (short)0, c, false, false);
}

// ---------------------------------------------------------------------------
// Phase 0: W_ih -> W_ihT [V,H] f32 (contiguous token-row gather),
// W_hh/W_out -> bf16 in [n][k] (contiguous-K B layout), bias sum, h0 -> bf16.
// ---------------------------------------------------------------------------
__global__ void prep_kernel(const float* __restrict__ W_ih,
                            const float* __restrict__ W_hh,
                            const float* __restrict__ W_out,
                            const float* __restrict__ b_ih,
                            const float* __restrict__ b_hh,
                            const float* __restrict__ h0,
                            float* __restrict__ wihT,
                            __bf16* __restrict__ whh_bf,
                            __bf16* __restrict__ wout_bf,
                            float* __restrict__ bias,
                            __bf16* __restrict__ hbf0) {
    const int stride = gridDim.x * blockDim.x;
    for (int i = blockIdx.x * blockDim.x + threadIdx.x; i < HIDDEN * VOCAB; i += stride) {
        const int h = i / VOCAB;
        const int v = i - h * VOCAB;
        wihT[v * HIDDEN + h] = W_ih[i];
        whh_bf[i]  = (__bf16)W_hh[i];
        wout_bf[i] = (__bf16)W_out[i];
        if (i < HIDDEN) bias[i] = b_ih[i] + b_hh[i];
        if (i < BATCH * HIDDEN) hbf0[i] = (__bf16)h0[i];
    }
}

// ---------------------------------------------------------------------------
// Phase 1 (x512, stream-ordered): h_t = tanh(W_ihT[tok] + bias + h@W_hh^T).
// Per block: TDM stages the block's 16x1024 bf16 W_hh slab into LDS once
// (padded 16B / 1024B so the 16 rows sit 8 banks apart); 4 waves (M-tiles)
// all consume it via ds_load_b128. ys (f32) written straight into d_out.
// ---------------------------------------------------------------------------
__global__ __launch_bounds__(128) void rnn_step_kernel(
        const __bf16* __restrict__ hin,     // [B,H] bf16 (h_{t-1})
        const __bf16* __restrict__ whh_bf,  // [H,H] bf16, [n][k]
        const float*  __restrict__ wihT,    // [V,H] f32
        const float*  __restrict__ bias,    // [H]
        const int*    __restrict__ x,       // [B,S] int32
        float*        __restrict__ ys,      // d_out: [S*B, H] f32
        __bf16*       __restrict__ hout,    // [B,H] bf16 (h_t)
        int s) {
    __shared__ __bf16 lds_b[16640];          // 16 rows * 2080 B (2048 + 2*16B pad)
    const int lane = threadIdx.x & 31;
    const int wave = threadIdx.x >> 5;       // 0..3 -> M tile
    const int n0   = blockIdx.x << 4;        // N tile

    if (wave == 0) {
        const unsigned long long ga =
            (unsigned long long)(uintptr_t)(whh_bf + (size_t)n0 * HIDDEN);
        const unsigned lds_base = (unsigned)(uintptr_t)&lds_b[0]; // LDS = flat[31:0]
        u32x4 g0 = { 1u,                                     // count=1 (valid D#)
                     lds_base,                               // lds_addr
                     (unsigned)ga,                           // global_addr[31:0]
                     (unsigned)((ga >> 32) & 0x1FFFFFFu)     // global_addr[56:32]
                         | (2u << 30) };                     // type=2 ("image")
        // dw0: data_size=1(2B) | pad_enable | pad_interval=7(256DW) | pad_amount=3(4DW)
        i32x8 g1 = { (int)0x07D10000,
                     (int)0x04000000,        // tensor_dim0 = 1024 (low16<<16)
                     (int)0x04000000,        // tensor_dim1 = 1024 (low16<<16)
                     (int)0x04000000,        // tile_dim0   = 1024 (<<16)
                     16,                     // tile_dim1   = 16 rows
                     1024,                   // tensor_dim0_stride = 1024 elems
                     0, 0 };
        i32x4 gz = { 0, 0, 0, 0 };
#if defined(__clang_major__) && __clang_major__ >= 23
        i32x8 gz8 = { 0, 0, 0, 0, 0, 0, 0, 0 };
        __builtin_amdgcn_tensor_load_to_lds(g0, g1, gz, gz, gz8, 0);
#else
        __builtin_amdgcn_tensor_load_to_lds(g0, g1, gz, gz, 0);
#endif
        __builtin_amdgcn_s_wait_tensorcnt(0);
    }
    __syncthreads();

    const int l15  = lane & 15;
    const int half = lane >> 4;
    const int m0   = wave << 4;
    const __bf16* arow_p = hin + (m0 + l15) * HIDDEN;   // A: row = M (global, L2-hot)
    const __bf16* brow_l = &lds_b[l15 * 1040];          // padded LDS row = 1040 bf16

    v8f acc = {};
    #pragma unroll 4
    for (int k0 = 0; k0 < HIDDEN; k0 += 32) {
        ABu a, b;
        const int akb = k0 + 8 * half;
        a.h[0] = *(const v8bf*)(arow_p + akb);
        a.h[1] = *(const v8bf*)(arow_p + akb + 16);
        const int bkb  = k0 + 8 * half;
        const int boff = bkb + 8 * (bkb >= 512);        // +16B pad past 1024B mark
        b.h[0] = *(const v8bf*)(brow_l + boff);         // ds_load_b128
        b.h[1] = *(const v8bf*)(brow_l + boff + 16);
        acc = wmma_bf16(a.v, b.v, acc);
    }

    const int ncol = n0 + l15;
    const float bn = bias[ncol];
    Accu ac; ac.v = acc;
    #pragma unroll
    for (int v = 0; v < 8; ++v) {
        const int brow = m0 + v + 8 * half;             // batch row (C: M = v+8*half)
        const int tok  = x[brow * SEQ + s];
        const float hv = tanhf(ac.f[v] + wihT[tok * HIDDEN + ncol] + bn);
        ys[((size_t)(s * BATCH + brow)) * HIDDEN + ncol] = hv;
        hout[brow * HIDDEN + ncol] = (__bf16)hv;
    }
}

// Save h_final (== ys[S-1]) before the in-place head GEMM destroys it.
__global__ void copy_hfinal_kernel(float* __restrict__ out) {
    const int i = blockIdx.x * blockDim.x + threadIdx.x;
    out[(size_t)SEQ * BATCH * HIDDEN + i] =
        out[(size_t)(SEQ - 1) * BATCH * HIDDEN + i];
}

// ---------------------------------------------------------------------------
// Phase 2: logits = ys @ W_out^T + b_out, IN PLACE over d_out (V == H).
// Each WG owns 32 rows staged as bf16 in 64 KB LDS; each wave register-blocks
// two N-tiles per A fragment (1 LDS A load -> 2 WMMAs).
// ---------------------------------------------------------------------------
__global__ __launch_bounds__(256) void head_gemm_kernel(
        float* __restrict__ out,                // [S*B, 1024] f32 (ys -> logits)
        const __bf16* __restrict__ wout_bf,     // [V,H] bf16, [n][k]
        const float*  __restrict__ b_out) {     // [V]
    __shared__ __bf16 lds_a[32 * HIDDEN];       // 64 KB
    const int row0 = blockIdx.x << 5;

    for (int i = threadIdx.x; i < 32 * HIDDEN; i += 256)
        lds_a[i] = (__bf16)out[(size_t)row0 * HIDDEN + i];
    __syncthreads();

    const int lane = threadIdx.x & 31;
    const int wave = threadIdx.x >> 5;          // 0..7
    const int l15  = lane & 15;
    const int half = lane >> 4;
    const int mt   = wave & 1;
    const __bf16* arow_p = &lds_a[(mt * 16 + l15) * HIDDEN];

    for (int np = (wave >> 1); np < 32; np += 4) {      // pairs of N-tiles
        const int ntA = 2 * np;
        const __bf16* browA = wout_bf + ((ntA * 16 + l15) * HIDDEN);
        const __bf16* browB = browA + 16 * HIDDEN;
        __builtin_prefetch(browA, 0, 0);                // global_prefetch_b8

        v8f acc0 = {}, acc1 = {};
        #pragma unroll 2
        for (int k0 = 0; k0 < HIDDEN; k0 += 32) {
            ABu a;
            const int akb = k0 + 8 * half;
            a.h[0] = *(const v8bf*)(arow_p + akb);      // ds_load_b128
            a.h[1] = *(const v8bf*)(arow_p + akb + 16);
            const int bkb = k0 + 8 * half;
            ABu b0, b1;
            b0.h[0] = *(const v8bf*)(browA + bkb);
            b0.h[1] = *(const v8bf*)(browA + bkb + 16);
            b1.h[0] = *(const v8bf*)(browB + bkb);
            b1.h[1] = *(const v8bf*)(browB + bkb + 16);
            acc0 = wmma_bf16(a.v, b0.v, acc0);
            acc1 = wmma_bf16(a.v, b1.v, acc1);
        }

        const int nc0 = ntA * 16 + l15, nc1 = nc0 + 16;
        const float bo0 = b_out[nc0], bo1 = b_out[nc1];
        Accu a0; a0.v = acc0;
        Accu a1; a1.v = acc1;
        #pragma unroll
        for (int v = 0; v < 8; ++v) {
            const int mrow = row0 + mt * 16 + v + 8 * half;
            out[(size_t)mrow * HIDDEN + nc0] = a0.f[v] + bo0;
            out[(size_t)mrow * HIDDEN + nc1] = a1.f[v] + bo1;
        }
    }
}

// ---------------------------------------------------------------------------
extern "C" void kernel_launch(void* const* d_in, const int* in_sizes, int n_in,
                              void* d_out, int out_size, void* d_ws, size_t ws_size,
                              hipStream_t stream) {
    (void)in_sizes; (void)n_in; (void)out_size; (void)ws_size;
    const int*   x     = (const int*)  d_in[0];
    const float* h0    = (const float*)d_in[1];
    const float* W_ih  = (const float*)d_in[2];
    const float* W_hh  = (const float*)d_in[3];
    const float* b_ih  = (const float*)d_in[4];
    const float* b_hh  = (const float*)d_in[5];
    const float* W_out = (const float*)d_in[6];
    const float* b_out = (const float*)d_in[7];
    float* out = (float*)d_out;

    char* ws = (char*)d_ws;                       // ~8.4 MB used
    float*  wihT    = (float*) (ws);                                  // 4 MB
    __bf16* whh_bf  = (__bf16*)(ws + (4u  << 20));                    // 2 MB
    __bf16* wout_bf = (__bf16*)(ws + (6u  << 20));                    // 2 MB
    float*  bias    = (float*) (ws + (8u  << 20));                    // 4 KB
    __bf16* hbf0    = (__bf16*)(ws + (8u  << 20) + 8192);             // 128 KB
    __bf16* hbf1    = hbf0 + BATCH * HIDDEN;                          // 128 KB

    prep_kernel<<<1024, 256, 0, stream>>>(W_ih, W_hh, W_out, b_ih, b_hh, h0,
                                          wihT, whh_bf, wout_bf, bias, hbf0);

    for (int s = 0; s < SEQ; ++s) {
        const __bf16* hin = (s & 1) ? hbf1 : hbf0;
        __bf16*      hout = (s & 1) ? hbf0 : hbf1;
        rnn_step_kernel<<<64, 128, 0, stream>>>(hin, whh_bf, wihT, bias,
                                                x, out, hout, s);
    }

    copy_hfinal_kernel<<<(BATCH * HIDDEN) / 256, 256, 0, stream>>>(out);
    head_gemm_kernel<<<(SEQ * BATCH) / 32, 256, 0, stream>>>(out, wout_bf, b_out);
}
